// GCN_22574348108245
// MI455X (gfx1250) — matrix operand order repaired and checked by reading, not compiled, not run on hardware
//
#include <hip/hip_runtime.h>
#include <math.h>

// GCN on MI455X (gfx1250): bf16 WMMA 16x16x32, fp32 accumulate.
// Reassociated: layer1 computed as (adj@x)@W1 (275+17 GFLOP instead of 17+550).
// All GEMMs: A row-major, B stored as B^T row-major, C stored transposed
// (contiguous 16B/lane). Wave tile = 32x64 (MT=2 x NT=4) -> 1.5 loads/WMMA.

typedef __attribute__((ext_vector_type(16))) __bf16 v16bf;
typedef __attribute__((ext_vector_type(8)))  float  v8f;

#define H_DIM  4096
#define W_FEAT 128
#define N_HID  256
#define BL_DIM 64

__device__ __forceinline__ unsigned short f2bf(float f) {
    unsigned int u = __float_as_uint(f);
    u += 0x7FFFu + ((u >> 16) & 1u);      // round-to-nearest-even
    return (unsigned short)(u >> 16);
}

union FragU { v16bf v; uint4 q[2]; };

// A frag (16x32 bf16) from row-major A at (m0,k0): lanes 0-15 row m0+l hold
// K {0..7,16..23}; lanes 16-31 hold K {8..15,24..31}.
__device__ __forceinline__ v16bf load_a(const unsigned short* A, int lda,
                                        int row, int kofs) {
    FragU a;
    const unsigned short* p = A + (size_t)row * lda + kofs;
    a.q[0] = *(const uint4*)(p);
    a.q[1] = *(const uint4*)(p + 16);
    return a.v;
}
// B frag (32x16 bf16) from B^T row-major: lane -> col, K {0..7,16..23} / {8..15,24..31}.
__device__ __forceinline__ v16bf load_b(const unsigned short* Bt, int ldbt,
                                        int col, int kofs) {
    FragU b;
    const unsigned short* p = Bt + (size_t)col * ldbt + kofs;
    b.q[0] = *(const uint4*)(p);
    b.q[1] = *(const uint4*)(p + 16);
    return b.v;
}

// ---------------------------------------------------------------------------
// Wave-tile GEMM:  C[M,N] = A[M,K] @ B[K,N]
//   A row-major bf16; Bt = B^T row-major bf16; Ct transposed store Ct[n][m].
// Each wave owns a (16*MT) x (16*NT) tile; B frags reused across MT.
// ---------------------------------------------------------------------------
template<int MT, int NT, bool BIAS_RELU>
__global__ __launch_bounds__(256)
void gemm_bf16_tn(const unsigned short* __restrict__ A,  int lda,  size_t sA,
                  const unsigned short* __restrict__ Bt, int ldbt, size_t sB,
                  unsigned short*       __restrict__ Ct, int ldct, size_t sC,
                  const float* __restrict__ bias,
                  int M, int K, int tilesPerBatch)
{
    const int wid  = blockIdx.x * (blockDim.x >> 5) + (threadIdx.x >> 5);
    const int lane = threadIdx.x & 31;
    const int b    = wid / tilesPerBatch;
    const int t    = wid - b * tilesPerBatch;
    const int TM   = M / (16 * MT);
    const int m0   = (t % TM) * (16 * MT);
    const int n0   = (t / TM) * (16 * NT);

    const unsigned short* Ab  = A  + (size_t)b * sA;
    const unsigned short* Btb = Bt + (size_t)b * sB;
    unsigned short*       Cb  = Ct + (size_t)b * sC;

    const int rowA = m0 + (lane & 15);
    const int kbA  = (lane >> 4) << 3;
    const int colB = n0 + (lane & 15);
    const int koB  = (lane >> 4) << 3;

    const v8f zero = {0.f,0.f,0.f,0.f,0.f,0.f,0.f,0.f};
    v8f acc[MT][NT];
#pragma unroll
    for (int i = 0; i < MT; ++i)
#pragma unroll
        for (int j = 0; j < NT; ++j) acc[i][j] = zero;

    for (int k0 = 0; k0 < K; k0 += 32) {
        v16bf a[MT];
#pragma unroll
        for (int i = 0; i < MT; ++i)
            a[i] = load_a(Ab, lda, rowA + 16 * i, k0 + kbA);
#pragma unroll
        for (int j = 0; j < NT; ++j) {
            const v16bf bf = load_b(Btb, ldbt, colB + 16 * j, k0 + koB);
#pragma unroll
            for (int i = 0; i < MT; ++i)
                acc[i][j] = __builtin_amdgcn_wmma_f32_16x16x32_bf16(
                    false, a[i], false, bf, (short)0, acc[i][j], false, false);
        }
    }

    // Transposed epilogue: lane holds column n, rows mbase..mbase+7 -> b128 store
    const int hi8 = (lane >> 4) << 3;
    float bv[MT][8];
    if (BIAS_RELU) {
#pragma unroll
        for (int i = 0; i < MT; ++i)
#pragma unroll
            for (int r = 0; r < 8; ++r)
                bv[i][r] = bias[m0 + 16 * i + hi8 + r];    // bias indexed by m
    }
#pragma unroll
    for (int i = 0; i < MT; ++i) {
        const int mbase = m0 + 16 * i + hi8;
#pragma unroll
        for (int j = 0; j < NT; ++j) {
            const int n = n0 + j * 16 + (lane & 15);
            union { unsigned short h[8]; uint4 q; } o;
#pragma unroll
            for (int r = 0; r < 8; ++r) {
                float v = acc[i][j][r];
                if (BIAS_RELU) v = fmaxf(v + bv[i][r], 0.0f);
                o.h[r] = f2bf(v);
            }
            *(uint4*)(Cb + (size_t)n * ldct + mbase) = o.q;
        }
    }
}

// ---------------------------------------------------------------------------
// Final layer: out = log_softmax(adj @ S2 + b2) over the 128 classes.
// Block = 8 waves covering one 32-row strip x all 128 columns (MT=2, B reused).
// ---------------------------------------------------------------------------
__global__ __launch_bounds__(256)
void gcn_out_logsoftmax(const unsigned short* __restrict__ adjb,
                        const unsigned short* __restrict__ S2t,
                        const float* __restrict__ b2,
                        float* __restrict__ out)
{
    __shared__ float tile[32][132];                // padded vs LDS bank conflicts
    const int b    = blockIdx.x >> 7;              // 128 row-strips per slice
    const int m0   = (blockIdx.x & 127) << 5;      // 32 rows per strip
    const int wv   = threadIdx.x >> 5;
    const int lane = threadIdx.x & 31;
    const int n0   = wv << 4;

    const unsigned short* Bb = S2t + (size_t)b * (size_t)W_FEAT * H_DIM;

    const int rowA = m0 + (lane & 15);
    const int kbA  = (lane >> 4) << 3;
    const int colB = n0 + (lane & 15);
    const int koB  = (lane >> 4) << 3;

    const v8f zero = {0.f,0.f,0.f,0.f,0.f,0.f,0.f,0.f};
    v8f acc0 = zero, acc1 = zero;
    for (int k0 = 0; k0 < H_DIM; k0 += 32) {
        const v16bf a0 = load_a(adjb, H_DIM, rowA,      k0 + kbA);
        const v16bf a1 = load_a(adjb, H_DIM, rowA + 16, k0 + kbA);
        const v16bf bf = load_b(Bb,   H_DIM, colB,      k0 + koB);
        acc0 = __builtin_amdgcn_wmma_f32_16x16x32_bf16(
            false, a0, false, bf, (short)0, acc0, false, false);
        acc1 = __builtin_amdgcn_wmma_f32_16x16x32_bf16(
            false, a1, false, bf, (short)0, acc1, false, false);
    }

    const int n  = n0 + (lane & 15);
    const float bias = b2[n];
    const int mb = (lane >> 4) << 3;
#pragma unroll
    for (int r = 0; r < 8; ++r) {
        tile[mb + r][n]      = acc0[r] + bias;
        tile[16 + mb + r][n] = acc1[r] + bias;
    }
    __syncthreads();

    // each wave reduces 4 rows of 128 (4 elems/lane + shuffle tree)
#pragma unroll
    for (int rr = 0; rr < 4; ++rr) {
        const int row = (wv << 2) + rr;
        float x0 = tile[row][lane];
        float x1 = tile[row][lane + 32];
        float x2 = tile[row][lane + 64];
        float x3 = tile[row][lane + 96];
        float mx = fmaxf(fmaxf(x0, x1), fmaxf(x2, x3));
#pragma unroll
        for (int off = 16; off > 0; off >>= 1)
            mx = fmaxf(mx, __shfl_xor(mx, off, 32));
        float s = expf(x0 - mx) + expf(x1 - mx) + expf(x2 - mx) + expf(x3 - mx);
#pragma unroll
        for (int off = 16; off > 0; off >>= 1)
            s += __shfl_xor(s, off, 32);
        const float lse = mx + logf(s);
        float* po = out + ((size_t)b * H_DIM + m0 + row) * W_FEAT;
        po[lane]      = x0 - lse;
        po[lane + 32] = x1 - lse;
        po[lane + 64] = x2 - lse;
        po[lane + 96] = x3 - lse;
    }
}

// ------------------------- prep kernels ------------------------------------
__global__ void cvt_f32_to_bf16_vec4(const float* __restrict__ src,
                                     unsigned short* __restrict__ dst, long n4)
{
    long i = (long)blockIdx.x * blockDim.x + threadIdx.x;
    if (i < n4) {
        float4 f = ((const float4*)src)[i];
        union { unsigned short h[4]; uint2 q; } o;
        o.h[0] = f2bf(f.x); o.h[1] = f2bf(f.y);
        o.h[2] = f2bf(f.z); o.h[3] = f2bf(f.w);
        ((uint2*)dst)[i] = o.q;
    }
}

// x[b][4096][128] f32 -> xt[b][128][4096] bf16 (32x32 LDS tile transpose)
__global__ void transpose_x_bf16(const float* __restrict__ x,
                                 unsigned short* __restrict__ xt)
{
    __shared__ float tile[32][33];
    const int bt = blockIdx.x;
    const int b  = bt >> 9;          // 512 tiles per slice (128 x 4)
    const int t  = bt & 511;
    const int r0 = (t >> 2) << 5;    // over H
    const int c0 = (t & 3) << 5;     // over W
    const float* xs = x + (size_t)b * H_DIM * W_FEAT;
#pragma unroll
    for (int i = 0; i < 32; i += 8)
        tile[threadIdx.y + i][threadIdx.x] =
            xs[(size_t)(r0 + threadIdx.y + i) * W_FEAT + c0 + threadIdx.x];
    __syncthreads();
    unsigned short* xd = xt + (size_t)b * W_FEAT * H_DIM;
#pragma unroll
    for (int i = 0; i < 32; i += 8)
        xd[(size_t)(c0 + threadIdx.y + i) * H_DIM + r0 + threadIdx.x] =
            f2bf(tile[threadIdx.x][threadIdx.y + i]);
}

// src[R][C] f32 -> dst[C][R] bf16 (tiny weight matrices)
__global__ void transpose_small_bf16(const float* __restrict__ src,
                                     unsigned short* __restrict__ dst, int R, int C)
{
    int i = blockIdx.x * blockDim.x + threadIdx.x;
    if (i < R * C) {
        int r = i / C, c = i - r * C;
        dst[(size_t)c * R + r] = f2bf(src[i]);
    }
}

// ---------------------------------------------------------------------------
extern "C" void kernel_launch(void* const* d_in, const int* in_sizes, int n_in,
                              void* d_out, int out_size, void* d_ws, size_t ws_size,
                              hipStream_t stream)
{
    (void)in_sizes; (void)n_in; (void)out_size; (void)ws_size;
    const float* x   = (const float*)d_in[0];
    const float* adj = (const float*)d_in[1];
    const float* W1  = (const float*)d_in[2];
    const float* b1  = (const float*)d_in[3];
    const float* W2  = (const float*)d_in[4];
    const float* b2  = (const float*)d_in[5];
    float* out = (float*)d_out;

    char* w = (char*)d_ws;
    size_t off = 0;
    auto carve = [&](size_t bytes) -> void* {
        void* p = w + off;
        off += (bytes + 255) & ~(size_t)255;
        return p;
    };
    unsigned short* adjb = (unsigned short*)carve((size_t)H_DIM * H_DIM * 2);           // 32 MB (L2-resident)
    unsigned short* W1t  = (unsigned short*)carve((size_t)N_HID * W_FEAT * 2);
    unsigned short* W2t  = (unsigned short*)carve((size_t)W_FEAT * N_HID * 2);
    unsigned short* Y    = (unsigned short*)carve((size_t)BL_DIM * H_DIM * W_FEAT * 2); // 64 MB
    unsigned short* Hbuf = (unsigned short*)carve((size_t)BL_DIM * H_DIM * N_HID * 2);  // 128 MB
    unsigned short* XT   = (unsigned short*)carve((size_t)BL_DIM * W_FEAT * H_DIM * 2); // 64 MB
    unsigned short* S2t  = XT;   // XT dead after GEMM1; reuse for S2^T

    // prep: bf16 conversions / transposes
    const long nadj4 = (long)H_DIM * H_DIM / 4;
    cvt_f32_to_bf16_vec4<<<(unsigned)((nadj4 + 255) / 256), 256, 0, stream>>>(adj, adjb, nadj4);
    transpose_small_bf16<<<(W_FEAT * N_HID + 255) / 256, 256, 0, stream>>>(W1, W1t, W_FEAT, N_HID);
    transpose_small_bf16<<<(N_HID * W_FEAT + 255) / 256, 256, 0, stream>>>(W2, W2t, N_HID, W_FEAT);
    transpose_x_bf16<<<BL_DIM * 512, dim3(32, 8), 0, stream>>>(x, XT);

    // GEMM1: Y^T = x^T @ adj^T  (M=128, N=4096, K=4096) -> Y row-major [4096][128]
    {
        const int M = W_FEAT, N = H_DIM, K = H_DIM;
        const int tiles = (M / 32) * (N / 64);            // MT=2, NT=4
        gemm_bf16_tn<2, 4, false><<<(tiles * BL_DIM) / 8, 256, 0, stream>>>(
            XT,   H_DIM,  (size_t)W_FEAT * H_DIM,
            adjb, H_DIM,  0,
            Y,    W_FEAT, (size_t)H_DIM * W_FEAT,
            nullptr, M, K, tiles);
    }
    // GEMM2: H^T = W1^T @ Y^T (+b1, relu)  (M=256, N=4096, K=128) -> H row-major
    {
        const int M = N_HID, N = H_DIM, K = W_FEAT;
        const int tiles = (M / 32) * (N / 64);
        gemm_bf16_tn<2, 4, true><<<(tiles * BL_DIM) / 8, 256, 0, stream>>>(
            W1t,  W_FEAT, 0,
            Y,    W_FEAT, (size_t)H_DIM * W_FEAT,
            Hbuf, N_HID,  (size_t)H_DIM * N_HID,
            b1, M, K, tiles);
    }
    // GEMM3: S2 = H @ W2  (M=4096, N=128, K=256) -> transposed store S2t[128][4096]
    {
        const int M = H_DIM, N = W_FEAT, K = N_HID;
        const int tiles = (M / 32) * (N / 64);
        gemm_bf16_tn<2, 4, false><<<(tiles * BL_DIM) / 8, 256, 0, stream>>>(
            Hbuf, N_HID, (size_t)H_DIM * N_HID,
            W2t,  N_HID, 0,
            S2t,  H_DIM, (size_t)W_FEAT * H_DIM,
            nullptr, M, K, tiles);
    }
    // GEMM4 + bias + log_softmax (32-row strips)
    gcn_out_logsoftmax<<<BL_DIM * (H_DIM / 32), 256, 0, stream>>>(adjb, S2t, b2, out);
}
